// SpatialNL_3882650435867
// MI455X (gfx1250) — matrix operand order repaired and checked by reading, not compiled, not run on hardware
//
#include <hip/hip_runtime.h>
#include <hip/hip_bf16.h>

// Spatial non-local block for MI455X (gfx1250, wave32, WMMA).
// LDS-staged, double-buffered WMMA GEMM using the CDNA5 async global->LDS
// copy path (GLOBAL_LOAD_ASYNC_TO_LDS_B128, ASYNCcnt).
// Workspace requirement: ~323 MB.

typedef __attribute__((ext_vector_type(16))) _Float16 v16h;
typedef __attribute__((ext_vector_type(8)))  _Float16 v8h;
typedef __attribute__((ext_vector_type(8)))  float    v8f;
typedef __attribute__((ext_vector_type(4)))  int      v4i;

constexpr int Bn = 4;      // batch
constexpr int Cc = 1024;   // channels
constexpr int Nn = 4096;   // H*W
constexpr int Pp = 512;    // projection dim
constexpr int M3 = 1536;   // 3*P (t,p,g stacked)

// LDS tile geometry: BM=BNt=128, BK=32, row padded to 40 halves (80B stride:
// 16 consecutive rows map to distinct bank quads for ds_load_b128).
constexpr int TILE_ROWS = 128;
constexpr int TILE_LDK  = 40;   // 32 + 8 pad halves
constexpr int TILE_HALF = TILE_ROWS * TILE_LDK;  // 5120 halves / tile

#if defined(__HIP_DEVICE_COMPILE__) && \
    __has_builtin(__builtin_amdgcn_global_load_async_to_lds_b128) && \
    __has_builtin(__builtin_amdgcn_s_wait_asynccnt)
#define USE_ASYNC_COPY 1
#else
#define USE_ASYNC_COPY 0
#endif

// 16-byte global -> LDS copy (per-lane). Async (ASYNCcnt) when supported.
__device__ __forceinline__ void copy16_g2l(const _Float16* g, _Float16* l) {
#if USE_ASYNC_COPY
    __builtin_amdgcn_global_load_async_to_lds_b128(
        (__attribute__((address_space(1))) v4i*)g,
        (__attribute__((address_space(3))) v4i*)l, 0, 0);
#else
    *(v8h*)l = *(const v8h*)g;
#endif
}

__device__ __forceinline__ void copy_wait() {
#if USE_ASYNC_COPY
    __builtin_amdgcn_s_wait_asynccnt(0);
#endif
}

// ---------------------------------------------------------------------------
// WMMA fragment helpers (CDNA5 wave32 layouts, 16-bit operands, K=32),
// reading from the padded LDS tiles.
// A 16x32 (MxK): lane row = m0+(lane&15); halves {8hi..8hi+7} u {16+8hi..}
// B 32x16 (KxN) from BT rows: lane col = n0+(lane&15); halves {8hi..8hi+7} u {16+8hi..}
// C/D 16x16: lane n fixed; element e -> m = 8*hi + e (contiguous in m)
// ---------------------------------------------------------------------------
__device__ __forceinline__ v16h lds_frag_a(const _Float16* sA, int row0, int lane) {
    int r  = row0 + (lane & 15);
    int hi = lane >> 4;
    const _Float16* base = sA + r * TILE_LDK + hi * 8;
    v8h lo = *(const v8h*)(base);
    v8h hh = *(const v8h*)(base + 16);
    v16h out;
#pragma unroll
    for (int i = 0; i < 8; ++i) { out[i] = lo[i]; out[i + 8] = hh[i]; }
    return out;
}

__device__ __forceinline__ v16h lds_frag_b(const _Float16* sB, int row0, int lane) {
    int r  = row0 + (lane & 15);
    int hi = lane >> 4;
    const _Float16* base = sB + r * TILE_LDK + hi * 8;
    v8h lo = *(const v8h*)(base);
    v8h hh = *(const v8h*)(base + 16);
    v16h out;
#pragma unroll
    for (int i = 0; i < 8; ++i) { out[i] = lo[i]; out[i + 8] = hh[i]; }
    return out;
}

// Transposed f16 store: OutT[n][m] row-major, one b128 per fragment per lane.
__device__ __forceinline__ void store_t(_Float16* __restrict__ OutT, int ldo,
                                        int m0, int n0, int lane, v8f c) {
    int n  = n0 + (lane & 15);
    int mb = m0 + 8 * (lane >> 4);
    v8h h;
#pragma unroll
    for (int e = 0; e < 8; ++e) h[e] = (_Float16)c[e];
    *(v8h*)(OutT + (size_t)n * ldo + mb) = h;
}

// Direct f32 store: Out[m][n] row-major (BN-friendly z layout).
__device__ __forceinline__ void store_d(float* __restrict__ Out, int ldo,
                                        int m0, int n0, int lane, v8f c) {
    int n  = n0 + (lane & 15);
    int mb = m0 + 8 * (lane >> 4);
#pragma unroll
    for (int e = 0; e < 8; ++e) Out[(size_t)(mb + e) * ldo + n] = c[e];
}

__device__ __forceinline__ v8f wmma_f16(v16h a, v16h b, v8f c) {
    return __builtin_amdgcn_wmma_f32_16x16x32_f16(
        false, a, false, b, (short)0, c, false, false);
}

// Stage one 128x32 A tile and one 128x32 BT tile into LDS (this thread's
// share: 4 x 16B chunks). 256 threads cover 2*512 chunks.
__device__ __forceinline__ void stage_tiles(
    const _Float16* __restrict__ A, const _Float16* __restrict__ BT,
    int lda, int ldb, int m0, int n0, int k,
    _Float16* sA, _Float16* sB, int t) {
#pragma unroll
    for (int j = 0; j < 2; ++j) {
        int id  = t + j * 256;      // 0..511
        int row = id >> 2;          // 0..127
        int ch  = (id & 3) * 8;     // halves 0,8,16,24
        copy16_g2l(A  + (size_t)(m0 + row) * lda + k + ch, sA + row * TILE_LDK + ch);
        copy16_g2l(BT + (size_t)(n0 + row) * ldb + k + ch, sB + row * TILE_LDK + ch);
    }
}

// ---------------------------------------------------------------------------
// TN GEMM: D(MxN) = A(MxK) * B(KxN); A row-major, BT row-major [N][K].
// Block: 256 threads = 8 waves (2m x 4n), block tile 128x128, BK=32,
// double-buffered LDS staging. Wave tile 64x32 (4x2 frags, 8 WMMA / k-step).
// ---------------------------------------------------------------------------
template <bool STORE_F32>
__global__ __launch_bounds__(256) void wmma_gemm_tn(
    const _Float16* __restrict__ A, const _Float16* __restrict__ BT,
    void* __restrict__ Out, int K, int lda, int ldb, int ldo,
    size_t strideA, size_t strideB, size_t strideO) {
    __shared__ _Float16 smem[2][2][TILE_HALF];  // [buf][A/B][tile] = 40 KB

    int bz = blockIdx.z;
    A  += strideA * bz;
    BT += strideB * bz;

    int t    = threadIdx.x;
    int lane = t & 31;
    int wave = t >> 5;
    int wm   = wave >> 2;             // 0..1 -> 64-row slab
    int wn   = wave & 3;              // 0..3 -> 32-col slab
    int m0   = blockIdx.y * 128;
    int n0   = blockIdx.x * 128;

    v8f acc[4][2] = {};
    int nt = K / 32;

    stage_tiles(A, BT, lda, ldb, m0, n0, 0, smem[0][0], smem[0][1], t);

    for (int kt = 0; kt < nt; ++kt) {
        int cur = kt & 1;
        copy_wait();
        __syncthreads();              // staged tile visible to all waves
        if (kt + 1 < nt) {            // prefetch next tile into other buffer
            stage_tiles(A, BT, lda, ldb, m0, n0, (kt + 1) * 32,
                        smem[cur ^ 1][0], smem[cur ^ 1][1], t);
        }
        const _Float16* sA = smem[cur][0];
        const _Float16* sB = smem[cur][1];
        v16h af[4], bf[2];
#pragma unroll
        for (int i = 0; i < 4; ++i) af[i] = lds_frag_a(sA, wm * 64 + i * 16, lane);
#pragma unroll
        for (int j = 0; j < 2; ++j) bf[j] = lds_frag_b(sB, wn * 32 + j * 16, lane);
#pragma unroll
        for (int i = 0; i < 4; ++i)
#pragma unroll
            for (int j = 0; j < 2; ++j)
                acc[i][j] = wmma_f16(af[i], bf[j], acc[i][j]);
        __syncthreads();              // all waves done reading buf[cur]
    }

    if (STORE_F32) {
        float* out = (float*)Out + strideO * bz;
#pragma unroll
        for (int i = 0; i < 4; ++i)
#pragma unroll
            for (int j = 0; j < 2; ++j)
                store_d(out, ldo, m0 + wm * 64 + i * 16,
                        n0 + wn * 32 + j * 16, lane, acc[i][j]);
    } else {
        _Float16* out = (_Float16*)Out + strideO * bz;
#pragma unroll
        for (int i = 0; i < 4; ++i)
#pragma unroll
            for (int j = 0; j < 2; ++j)
                store_t(out, ldo, m0 + wm * 64 + i * 16,
                        n0 + wn * 32 + j * 16, lane, acc[i][j]);
    }
}

// ---------------------------------------------------------------------------
// Weight conversion: Wall[1536][1024] f16 from Wt/Wp/Wg, Wz16[1024][512] f16.
// ---------------------------------------------------------------------------
__global__ __launch_bounds__(256) void k_cvt_w(
    const float* __restrict__ Wt, const float* __restrict__ Wp,
    const float* __restrict__ Wg, const float* __restrict__ Wz,
    _Float16* __restrict__ Wall, _Float16* __restrict__ Wz16) {
    int idx = blockIdx.x * 256 + threadIdx.x;
    const int seg = Pp * Cc;  // 524288
    if (idx < seg) {
        Wall[idx] = (_Float16)Wt[idx];
    } else if (idx < 2 * seg) {
        Wall[idx] = (_Float16)Wp[idx - seg];
    } else if (idx < 3 * seg) {
        Wall[idx] = (_Float16)Wg[idx - 2 * seg];
    } else if (idx < 4 * seg) {
        Wz16[idx - 3 * seg] = (_Float16)Wz[idx - 3 * seg];
    }
}

// ---------------------------------------------------------------------------
// x [b][C][N] f32  ->  xT [b][N][C] f16  (LDS tile transpose + convert)
// ---------------------------------------------------------------------------
__global__ __launch_bounds__(256) void k_x_transpose(
    const float* __restrict__ x, _Float16* __restrict__ xT) {
    __shared__ _Float16 tile[32][33];
    int b  = blockIdx.z;
    int n0 = blockIdx.x * 32;
    int c0 = blockIdx.y * 32;
    int tx = threadIdx.x, ty = threadIdx.y;  // (32, 8)
    const float* xin = x + ((size_t)b * Cc + c0) * Nn + n0;
#pragma unroll
    for (int i = 0; i < 4; ++i) {
        int c = ty + i * 8;
        tile[c][tx] = (_Float16)xin[(size_t)c * Nn + tx];  // tile[c][n]
    }
    __syncthreads();
    _Float16* outp = xT + ((size_t)b * Nn + n0) * Cc + c0;
#pragma unroll
    for (int i = 0; i < 4; ++i) {
        int n = ty + i * 8;
        outp[(size_t)n * Cc + tx] = tile[tx][n];           // out[n][c]
    }
}

// ---------------------------------------------------------------------------
// gT (cols 1024..1535 of tpgT [b][N][1536]) -> g_pn [b][512][N] f16
// ---------------------------------------------------------------------------
__global__ __launch_bounds__(256) void k_g_transpose(
    const _Float16* __restrict__ tpgT, _Float16* __restrict__ g_pn) {
    __shared__ _Float16 tile[32][33];
    int b  = blockIdx.z;
    int n0 = blockIdx.x * 32;
    int p0 = blockIdx.y * 32;
    int tx = threadIdx.x, ty = threadIdx.y;  // (32, 8)
#pragma unroll
    for (int i = 0; i < 4; ++i) {
        int n = ty + i * 8;
        tile[n][tx] = tpgT[((size_t)b * Nn + n0 + n) * M3 + 1024 + p0 + tx];
    }
    __syncthreads();
#pragma unroll
    for (int i = 0; i < 4; ++i) {
        int p = ty + i * 8;
        g_pn[((size_t)b * Pp + p0 + p) * Nn + n0 + tx] = tile[tx][p];
    }
}

// ---------------------------------------------------------------------------
// Row softmax over j: att [B*N rows][4096 f16], one block per row.
// ---------------------------------------------------------------------------
__global__ __launch_bounds__(256) void k_softmax(_Float16* __restrict__ att) {
    size_t row = blockIdx.x;
    _Float16* r = att + row * (size_t)Nn;
    int t = threadIdx.x;
    float vals[16];
    float mx = -3.0e38f;
#pragma unroll
    for (int i = 0; i < 16; ++i) {
        vals[i] = (float)r[i * 256 + t];
        mx = fmaxf(mx, vals[i]);
    }
    __shared__ float red[256];
    red[t] = mx;
    __syncthreads();
    for (int s = 128; s > 0; s >>= 1) {
        if (t < s) red[t] = fmaxf(red[t], red[t + s]);
        __syncthreads();
    }
    mx = red[0];
    __syncthreads();
    float sum = 0.f;
#pragma unroll
    for (int i = 0; i < 16; ++i) {
        vals[i] = __expf(vals[i] - mx);
        sum += vals[i];
    }
    red[t] = sum;
    __syncthreads();
    for (int s = 128; s > 0; s >>= 1) {
        if (t < s) red[t] += red[t + s];
        __syncthreads();
    }
    float inv = 1.0f / red[0];
#pragma unroll
    for (int i = 0; i < 16; ++i) r[i * 256 + t] = (_Float16)(vals[i] * inv);
}

// ---------------------------------------------------------------------------
// BatchNorm: per-channel mean / invstd over (b, n). One block per channel.
// ---------------------------------------------------------------------------
__global__ __launch_bounds__(256) void k_bn_reduce(
    const float* __restrict__ z, float* __restrict__ mean,
    float* __restrict__ invstd) {
    int c = blockIdx.x;
    int t = threadIdx.x;
    float s = 0.f, sq = 0.f;
    for (int b = 0; b < Bn; ++b) {
        const float* p = z + ((size_t)b * Cc + c) * Nn;
        for (int i = t; i < Nn; i += 256) {
            float v = p[i];
            s += v;
            sq += v * v;
        }
    }
    __shared__ float rs[256], rq[256];
    rs[t] = s; rq[t] = sq;
    __syncthreads();
    for (int st = 128; st > 0; st >>= 1) {
        if (t < st) { rs[t] += rs[t + st]; rq[t] += rq[t + st]; }
        __syncthreads();
    }
    if (t == 0) {
        const float invN = 1.0f / (float)(Bn * Nn);
        float m   = rs[0] * invN;
        float var = rq[0] * invN - m * m;
        mean[c]   = m;
        invstd[c] = rsqrtf(var + 1e-5f);
    }
}

__global__ __launch_bounds__(256) void k_bn_apply(
    const float* __restrict__ z, const float* __restrict__ x,
    const float* __restrict__ mean, const float* __restrict__ invstd,
    const float* __restrict__ gamma, const float* __restrict__ beta,
    float* __restrict__ out) {
    size_t idx = (size_t)blockIdx.x * 256 + threadIdx.x;  // B*C*N total
    int c = (int)((idx / (size_t)Nn) % (size_t)Cc);
    out[idx] = (z[idx] - mean[c]) * invstd[c] * gamma[c] + beta[c] + x[idx];
}

// ---------------------------------------------------------------------------
extern "C" void kernel_launch(void* const* d_in, const int* in_sizes, int n_in,
                              void* d_out, int out_size, void* d_ws, size_t ws_size,
                              hipStream_t stream) {
    const float* x     = (const float*)d_in[0];
    const float* Wt    = (const float*)d_in[1];
    const float* Wp    = (const float*)d_in[2];
    const float* Wg    = (const float*)d_in[3];
    const float* Wz    = (const float*)d_in[4];
    const float* gamma = (const float*)d_in[5];
    const float* beta  = (const float*)d_in[6];
    float* out = (float*)d_out;

    char* ws = (char*)d_ws;
    // workspace layout (all offsets 16B aligned)
    size_t o_wall = 0;                                       // 1536*1024 f16
    size_t o_wz   = o_wall + (size_t)M3 * Cc * 2;            // 1024*512  f16
    size_t o_xT   = o_wz   + (size_t)Cc * Pp * 2;            // B*N*C     f16
    size_t o_tpg  = o_xT   + (size_t)Bn * Nn * Cc * 2;       // B*N*1536  f16
    size_t o_g    = o_tpg  + (size_t)Bn * Nn * M3 * 2;       // B*512*N   f16
    size_t o_att  = o_g    + (size_t)Bn * Pp * Nn * 2;       // B*N*N     f16
    size_t o_y    = o_att  + (size_t)Bn * Nn * Nn * 2;       // B*N*512   f16
    size_t o_z    = o_y    + (size_t)Bn * Nn * Pp * 2;       // B*C*N     f32
    size_t o_bn   = o_z    + (size_t)Bn * Cc * Nn * 4;       // 2*C       f32

    _Float16* Wall = (_Float16*)(ws + o_wall);
    _Float16* Wz16 = (_Float16*)(ws + o_wz);
    _Float16* xT   = (_Float16*)(ws + o_xT);
    _Float16* tpgT = (_Float16*)(ws + o_tpg);
    _Float16* g_pn = (_Float16*)(ws + o_g);
    _Float16* att  = (_Float16*)(ws + o_att);
    _Float16* yT   = (_Float16*)(ws + o_y);
    float*    z    = (float*)(ws + o_z);
    float*    mean = (float*)(ws + o_bn);
    float*    istd = mean + Cc;

    // 1) convert weights to f16 (Wall = [Wt;Wp;Wg])
    k_cvt_w<<<(4 * Pp * Cc) / 256, 256, 0, stream>>>(Wt, Wp, Wg, Wz, Wall, Wz16);

    // 2) xT[b][n][c] = x[b][c][n]  (f16)
    k_x_transpose<<<dim3(Nn / 32, Cc / 32, Bn), dim3(32, 8), 0, stream>>>(x, xT);

    // 3) GEMM1: tpgT[b][n][m] = (Wall(1536x1024) * x(1024x4096))^T
    wmma_gemm_tn<false><<<dim3(Nn / 128, M3 / 128, Bn), 256, 0, stream>>>(
        Wall, xT, tpgT, /*K=*/Cc, /*lda=*/Cc, /*ldb=*/Cc, /*ldo=*/M3,
        0, (size_t)Nn * Cc, (size_t)Nn * M3);

    // 4) GEMM2: logits. A = pT (cols 512..1023), BT = tT (cols 0..511);
    //    D[m=j][n=i] stored transposed -> att[i][j] row-major f16.
    wmma_gemm_tn<false><<<dim3(Nn / 128, Nn / 128, Bn), 256, 0, stream>>>(
        tpgT + Pp, tpgT, att, /*K=*/Pp, /*lda=*/M3, /*ldb=*/M3, /*ldo=*/Nn,
        (size_t)Nn * M3, (size_t)Nn * M3, (size_t)Nn * Nn);

    // 5) softmax over j for each of B*N rows
    k_softmax<<<Bn * Nn, 256, 0, stream>>>(att);

    // 6) g_pn[b][p][n] from tpgT cols 1024..1535
    k_g_transpose<<<dim3(Nn / 32, Pp / 32, Bn), dim3(32, 8), 0, stream>>>(tpgT, g_pn);

    // 7) GEMM3: y. A = g(512x4096), BT = att[i][j]; D[m=p][n=i] transposed
    //    -> yT[b][i][p] f16.
    wmma_gemm_tn<false><<<dim3(Nn / 128, Pp / 128, Bn), 256, 0, stream>>>(
        g_pn, att, yT, /*K=*/Nn, /*lda=*/Nn, /*ldb=*/Nn, /*ldo=*/Pp,
        (size_t)Pp * Nn, (size_t)Nn * Nn, (size_t)Nn * Pp);

    // 8) GEMM4: z[b][c][n] f32 direct. A = Wz16(1024x512), BT = yT.
    wmma_gemm_tn<true><<<dim3(Nn / 128, Cc / 128, Bn), 256, 0, stream>>>(
        Wz16, yT, z, /*K=*/Pp, /*lda=*/Pp, /*ldb=*/Pp, /*ldo=*/Nn,
        0, (size_t)Nn * Pp, (size_t)Cc * Nn);

    // 9) BN statistics per channel
    k_bn_reduce<<<Cc, 256, 0, stream>>>(z, mean, istd);

    // 10) BN apply + residual
    k_bn_apply<<<(Bn * Cc * Nn) / 256, 256, 0, stream>>>(
        z, x, mean, istd, gamma, beta, out);
}
